// MultiHeadSelfAttention_58686433133210
// MI455X (gfx1250) — matrix operand order repaired and checked by reading, not compiled, not run on hardware
//
#include <hip/hip_runtime.h>
#include <hip/hip_bf16.h>
#include <math.h>

// ---------------------------------------------------------------------------
// MI455X (gfx1250) multi-head self-attention with RoPE, bf16 WMMA pipeline.
// Compute-bound (~137 GFLOP vs <10us of HBM traffic at 23.3 TB/s): all matmuls
// on V_WMMA_F32_16X16X32_BF16 (f32 accumulate).  192MB L2 holds every
// intermediate.  Attention stages K/V tiles into LDS with the Tensor Data
// Mover (tensor_load_to_lds, double-buffered, s_wait_tensorcnt) so DMA of
// block i+1 overlaps WMMA+softmax of block i.
// ---------------------------------------------------------------------------

#define D_MODEL   1024
#define NUM_HEADS 16
#define D_KH      64
#define SEQ       2048
#define BATCH     4
#define MTOT      (BATCH * SEQ)   // 8192 rows

typedef __bf16 bf16;
typedef __attribute__((ext_vector_type(8)))  __bf16 v8bf;
typedef __attribute__((ext_vector_type(16))) __bf16 v16bf;
typedef __attribute__((ext_vector_type(8)))  float  v8f;
typedef __attribute__((ext_vector_type(4)))  unsigned int v4u;
typedef __attribute__((ext_vector_type(8)))  unsigned int v8u;

static __device__ __forceinline__ v16bf cat8(v8bf lo, v8bf hi) {
  return __builtin_shufflevector(lo, hi, 0,1,2,3,4,5,6,7,8,9,10,11,12,13,14,15);
}

static __device__ __forceinline__ v8f wmma_bf16(v16bf a, v16bf b, v8f c) {
  // (neg_a, A, neg_b, B, c_mod, C, reuse_a, reuse_b)
  return __builtin_amdgcn_wmma_f32_16x16x32_bf16(false, a, false, b, (short)0, c,
                                                 false, false);
}

// ---------------------------------------------------------------------------
// TDM: issue a 2D tensor tile load (global -> LDS).  D# per ISA 08 §8.3/8.4:
//   group0: count=1, lds_addr, 57b global_addr, type=2
//   group1: data_size=1 (2B), tensor_dim0/1, tile_dim0/1, dim0 stride
// 2D tensors use the 2-SGPR-group form (VADDR2/3 = NULL).  Tracked by
// TENSORcnt.  All inputs are wave-uniform.
// ---------------------------------------------------------------------------
static __device__ __forceinline__ void tdm_load_2d(unsigned lds_off,
                                                   const void* gaddr,
                                                   unsigned tensor_d0,
                                                   unsigned tensor_d1,
                                                   unsigned tile_d0,
                                                   unsigned tile_d1,
                                                   unsigned stride0_elems) {
  unsigned long long ga = (unsigned long long)(uintptr_t)gaddr;
  v4u g0;
  g0[0] = 1u;                                               // count=1
  g0[1] = lds_off;                                          // LDS byte offset
  g0[2] = (unsigned)ga;                                     // global_addr lo
  g0[3] = (unsigned)((ga >> 32) & 0x01FFFFFFu) | (2u << 30);// addr hi | type=2
  v8u g1;
  g1[0] = 1u << 16;                                         // data_size=1 (2B)
  g1[1] = (tensor_d0 & 0xFFFFu) << 16;                      // tensor_dim0 lo16
  g1[2] = (tensor_d0 >> 16) | ((tensor_d1 & 0xFFFFu) << 16);
  g1[3] = (tensor_d1 >> 16) | (tile_d0 << 16);              // tile_dim0
  g1[4] = tile_d1;                                          // tile_dim1 (d2=0)
  g1[5] = stride0_elems;                                    // dim0 stride lo32
  g1[6] = 0u;                                               // stride hi / d1 stride
  g1[7] = 0u;
  asm volatile("tensor_load_to_lds %0, %1" :: "s"(g0), "s"(g1) : "memory");
}

// ---------------------------------------------------------------------------
// fp32 -> bf16 elementwise convert
// ---------------------------------------------------------------------------
__global__ void cvt_f32_bf16(const float* __restrict__ in, bf16* __restrict__ out,
                             int n) {
  int i = blockIdx.x * blockDim.x + threadIdx.x;
  if (i < n) out[i] = (bf16)in[i];
}

// ---------------------------------------------------------------------------
// C[m,n] = sum_k A[m,k] * B[n,k]   (y = x @ W^T, W stored [out,in]).
// Per-wave tile 32(M) x 64(N); 8 WMMAs per K-step of 32; operands stream
// through the 192MB L2 with global_prefetch_b8 lookahead.
// ---------------------------------------------------------------------------
__global__ __launch_bounds__(256) void gemm_nt_bf16(const bf16* __restrict__ A,
                                                    const bf16* __restrict__ Bm,
                                                    float* __restrict__ C,
                                                    int M, int N, int K) {
  const int lane = threadIdx.x & 31;
  const int wid  = blockIdx.x * (blockDim.x >> 5) + (threadIdx.x >> 5);
  const int tilesN = N >> 6;
  const int mt = wid / tilesN, nt = wid % tilesN;
  const int mb = mt << 5, nb = nt << 6;
  if (mb >= M) return;
  const int ln = lane & 15;
  const int hf = lane >> 4;
  const int koffA = hf ? 16 : 0;
  const int koffB = hf ? 16 : 0;

  const v8f vzero = {0.f, 0.f, 0.f, 0.f, 0.f, 0.f, 0.f, 0.f};
  v8f acc[2][4];
#pragma unroll
  for (int i = 0; i < 2; ++i)
#pragma unroll
    for (int j = 0; j < 4; ++j) acc[i][j] = vzero;

  const bf16* aptr0 = A + (size_t)(mb + ln) * K;
  const bf16* aptr1 = A + (size_t)(mb + 16 + ln) * K;
  const bf16* bptr[4];
#pragma unroll
  for (int j = 0; j < 4; ++j)
    bptr[j] = Bm + (size_t)(nb + 16 * j + ln) * K + koffB;

  for (int k0 = 0; k0 < K; k0 += 32) {
    __builtin_prefetch(aptr0 + k0 + 256, 0, 1);  // global_prefetch_b8
    __builtin_prefetch(bptr[0] + k0 + 256, 0, 1);
    v16bf af0 = cat8(*(const v8bf*)(aptr0 + k0 + koffA),
                     *(const v8bf*)(aptr0 + k0 + koffA + 8));
    v16bf af1 = cat8(*(const v8bf*)(aptr1 + k0 + koffA),
                     *(const v8bf*)(aptr1 + k0 + koffA + 8));
#pragma unroll
    for (int j = 0; j < 4; ++j) {
      v16bf bfr = *(const v16bf*)(bptr[j] + k0);   // 32B contiguous
      acc[0][j] = wmma_bf16(af0, bfr, acc[0][j]);
      acc[1][j] = wmma_bf16(af1, bfr, acc[1][j]);
    }
  }

#pragma unroll
  for (int i = 0; i < 2; ++i)
#pragma unroll
    for (int j = 0; j < 4; ++j) {
      int n = nb + 16 * j + ln;
#pragma unroll
      for (int v = 0; v < 8; ++v) {
        int m = mb + 16 * i + v + 8 * hf;
        C[(size_t)m * N + n] = acc[i][j][v];
      }
    }
}

// ---------------------------------------------------------------------------
// RoPE + repack: fp32 [B*S, 1024] -> bf16 [B, H, S, 64]
// ---------------------------------------------------------------------------
__global__ void rope_repack(const float* __restrict__ src, bf16* __restrict__ dst) {
  int idx = blockIdx.x * blockDim.x + threadIdx.x;   // B*S*H*32 pairs
  if (idx >= MTOT * NUM_HEADS * (D_KH / 2)) return;
  int j = idx & 31;
  int h = (idx >> 5) & (NUM_HEADS - 1);
  int m = idx >> 9;              // b*S + s
  int s = m & (SEQ - 1);
  int b = m >> 11;
  size_t sbase = (size_t)m * D_MODEL + h * D_KH + 2 * j;
  float e0 = src[sbase];
  float e1 = src[sbase + 1];
  float freq = __powf(10000.0f, -(2.0f * (float)j) / 64.0f);
  float ang = (float)s * freq;
  float sn, cs;
  __sincosf(ang, &sn, &cs);
  size_t dbase = (((size_t)(b * NUM_HEADS + h) * SEQ) + s) * D_KH + 2 * j;
  dst[dbase]     = (bf16)(cs * e0 - sn * e1);
  dst[dbase + 1] = (bf16)(sn * e0 + cs * e1);
}

// ---------------------------------------------------------------------------
// V repack + transpose: fp32 [B*S, 1024] -> bf16 [B, H, 64, S]
// ---------------------------------------------------------------------------
__global__ void v_repack(const float* __restrict__ src, bf16* __restrict__ dst) {
  int idx = blockIdx.x * blockDim.x + threadIdx.x;   // MTOT * D_MODEL
  if (idx >= MTOT * D_MODEL) return;
  int c = idx & (D_MODEL - 1);
  int m = idx >> 10;
  int h = c >> 6, d = c & 63;
  int s = m & (SEQ - 1), b = m >> 11;
  dst[(((size_t)(b * NUM_HEADS + h) * D_KH + d) * SEQ) + s] = (bf16)src[idx];
}

// ---------------------------------------------------------------------------
// Flash attention, causal.  One wave per (b, h, 32-query tile).
// K/V tiles (32 keys) are TDM'd into double-buffered LDS; block i+1's DMA
// overlaps block i's WMMAs.  Score C-layout: lane n = key (L%16), row m =
// vgpr + 8*(L>=16); row stats reduced with __shfl_xor {1,2,4,8} (intra-half).
// P transposed to WMMA A-layout through LDS.
// ---------------------------------------------------------------------------
#define BIG_NEG (-1e30f)

__global__ __launch_bounds__(32) void attn_flash(const bf16* __restrict__ Q,
                                                 const bf16* __restrict__ Kk,
                                                 const bf16* __restrict__ Vt,
                                                 bf16* __restrict__ Ob) {
  __shared__ __align__(16) bf16 sK[2][32 * 64];   // [buf][key][d]     4KB ea
  __shared__ __align__(16) bf16 sV[2][64 * 32];   // [buf][d][key]     4KB ea
  __shared__ __align__(16) bf16 sP[32 * 40];      // 32 q-rows x 32 keys (pad)

  const int lane = threadIdx.x;
  const int ln = lane & 15;
  const int hf = lane >> 4;
  const int bid = blockIdx.x;                 // B*H*(S/32) = 4096
  const int qt  = bid & ((SEQ / 32) - 1);     // 0..63
  const int bh  = bid >> 6;                   // 0..63  (b*H + h)
  const int q_base = qt * 32;
  const int b = bh >> 4, h = bh & 15;

  const bf16* Qh = Q  + (size_t)bh * SEQ * D_KH;
  const bf16* Kh = Kk + (size_t)bh * SEQ * D_KH;
  const bf16* Vh = Vt + (size_t)bh * D_KH * SEQ;

  const unsigned sKoff0 = (unsigned)(uintptr_t)&sK[0][0];
  const unsigned sKoff1 = (unsigned)(uintptr_t)&sK[1][0];
  const unsigned sVoff0 = (unsigned)(uintptr_t)&sV[0][0];
  const unsigned sVoff1 = (unsigned)(uintptr_t)&sV[1][0];

  // --- load the 32x64 Q tile as four A-fragments (held for the whole pass)
  v16bf qf[2][2];
  const int koffA = hf ? 16 : 0;
#pragma unroll
  for (int u = 0; u < 2; ++u) {
    const bf16* p = Qh + (size_t)(q_base + 16 * u + ln) * D_KH;
#pragma unroll
    for (int t = 0; t < 2; ++t)
      qf[u][t] = cat8(*(const v8bf*)(p + t * 32 + koffA),
                      *(const v8bf*)(p + t * 32 + koffA + 8));
  }

  const v8f vzero = {0.f, 0.f, 0.f, 0.f, 0.f, 0.f, 0.f, 0.f};
  v8f acc[2][4];
  float m_i[2][8], l_i[2][8];
#pragma unroll
  for (int u = 0; u < 2; ++u) {
#pragma unroll
    for (int j = 0; j < 4; ++j) acc[u][j] = vzero;
#pragma unroll
    for (int v = 0; v < 8; ++v) { m_i[u][v] = BIG_NEG; l_i[u][v] = 0.f; }
  }

  const int kend = q_base + 32;               // causal bound; multiple of 32

  // --- TDM prefetch of key-block 0 (K: 32x64 tile, V^T: 64x32 tile)
  tdm_load_2d(sKoff0, Kh,     /*td0*/ D_KH, /*td1*/ SEQ, /*tile*/ D_KH, 32, D_KH);
  tdm_load_2d(sVoff0, Vh,     /*td0*/ SEQ,  /*td1*/ D_KH, /*tile*/ 32, D_KH, SEQ);

  int bufi = 0;
  for (int kb = 0; kb < kend; kb += 32, bufi ^= 1) {
    if (kb + 32 < kend) {   // kick off next block's DMA, then wait for current
      tdm_load_2d(bufi ? sKoff0 : sKoff1, Kh + (size_t)(kb + 32) * D_KH,
                  D_KH, SEQ, D_KH, 32, D_KH);
      tdm_load_2d(bufi ? sVoff0 : sVoff1, Vh + (kb + 32),
                  SEQ, D_KH, 32, D_KH, SEQ);
      asm volatile("s_wait_tensorcnt 0x2" ::: "memory");
    } else {
      asm volatile("s_wait_tensorcnt 0x0" ::: "memory");
    }
    const bf16* kbuf = &sK[bufi][0];
    const bf16* vbuf = &sV[bufi][0];

    // --- scores S = Q K^T : 32q x 32k block (2 q-tiles x 2 key-tiles)
    v8f st[2][2];
#pragma unroll
    for (int t = 0; t < 2; ++t) {
      const bf16* kp = kbuf + (t * 16 + ln) * D_KH + (hf ? 16 : 0);
      v16bf b0 = *(const v16bf*)(kp);        // d 0..31  (lane's 16)
      v16bf b1 = *(const v16bf*)(kp + 32);   // d 32..63
#pragma unroll
      for (int u = 0; u < 2; ++u) {
        v8f s = vzero;
        s = wmma_bf16(qf[u][0], b0, s);
        s = wmma_bf16(qf[u][1], b1, s);
        st[u][t] = s;
      }
    }

    // --- online softmax
#pragma unroll
    for (int u = 0; u < 2; ++u) {
#pragma unroll
      for (int v = 0; v < 8; ++v) {
        const int q = q_base + 16 * u + v + 8 * hf;
        float s0 = st[u][0][v] * 0.125f;     // 1/sqrt(64)
        float s1 = st[u][1][v] * 0.125f;
        if (kb + ln > q)      s0 = BIG_NEG;  // causal mask
        if (kb + 16 + ln > q) s1 = BIG_NEG;
        float mx = fmaxf(s0, s1);
#pragma unroll
        for (int off = 1; off < 16; off <<= 1)
          mx = fmaxf(mx, __shfl_xor(mx, off, 32));
        const float nm = fmaxf(m_i[u][v], mx);
        const float alpha = __expf(m_i[u][v] - nm);
        const float p0 = __expf(s0 - nm);
        const float p1 = __expf(s1 - nm);
        float ps = p0 + p1;
#pragma unroll
        for (int off = 1; off < 16; off <<= 1)
          ps += __shfl_xor(ps, off, 32);
        l_i[u][v] = l_i[u][v] * alpha + ps;
        m_i[u][v] = nm;
#pragma unroll
        for (int j = 0; j < 4; ++j) acc[u][j][v] *= alpha;
        sP[(16 * u + v + 8 * hf) * 40 + ln]      = (bf16)p0;
        sP[(16 * u + v + 8 * hf) * 40 + 16 + ln] = (bf16)p1;
      }
    }
    __syncthreads();

    // --- reload P as two 16x32 A-fragments
    v16bf pf[2];
#pragma unroll
    for (int u = 0; u < 2; ++u) {
      const bf16* pp = &sP[(16 * u + ln) * 40];
      pf[u] = cat8(*(const v8bf*)(pp + koffA),
                   *(const v8bf*)(pp + koffA + 8));
    }

    // --- O += P V  (V-fragments shared across both q-tiles)
#pragma unroll
    for (int j = 0; j < 4; ++j) {
      v16bf vfr = *(const v16bf*)(vbuf + (j * 16 + ln) * 32 + (hf ? 16 : 0));
      acc[0][j] = wmma_bf16(pf[0], vfr, acc[0][j]);
      acc[1][j] = wmma_bf16(pf[1], vfr, acc[1][j]);
    }
    __syncthreads();
  }

  // --- normalize and write bf16 [B*S, H*64] for the output projection
#pragma unroll
  for (int u = 0; u < 2; ++u)
#pragma unroll
    for (int v = 0; v < 8; ++v) {
      const float inv = 1.0f / l_i[u][v];
      const int q = q_base + 16 * u + v + 8 * hf;
      const size_t row = (size_t)(b * SEQ + q) * D_MODEL + h * D_KH;
#pragma unroll
      for (int j = 0; j < 4; ++j)
        Ob[row + j * 16 + ln] = (bf16)(acc[u][j][v] * inv);
    }
}

// ---------------------------------------------------------------------------
// host-side orchestration
// ---------------------------------------------------------------------------
extern "C" void kernel_launch(void* const* d_in, const int* in_sizes, int n_in,
                              void* d_out, int out_size, void* d_ws, size_t ws_size,
                              hipStream_t stream) {
  const float* X  = (const float*)d_in[0];
  const float* wq = (const float*)d_in[1];
  const float* wk = (const float*)d_in[2];
  const float* wv = (const float*)d_in[3];
  const float* wo = (const float*)d_in[4];
  float* out = (float*)d_out;

  char* p = (char*)d_ws;
  auto carve = [&](size_t bytes) {
    char* r = p;
    p += (bytes + 255) & ~(size_t)255;
    return r;
  };
  const size_t nX = (size_t)MTOT * D_MODEL;       // 8.4M elems
  const size_t nW = (size_t)D_MODEL * D_MODEL;    // 1.05M elems

  bf16*  Xb   = (bf16*)carve(nX * 2);
  bf16*  Wqb  = (bf16*)carve(nW * 2);
  bf16*  Wkb  = (bf16*)carve(nW * 2);
  bf16*  Wvb  = (bf16*)carve(nW * 2);
  bf16*  Wob  = (bf16*)carve(nW * 2);
  float* tmpF = (float*)carve(nX * 4);            // reused fp32 staging
  bf16*  Qb   = (bf16*)carve(nX * 2);             // [B,H,S,64]
  bf16*  Kb   = (bf16*)carve(nX * 2);             // [B,H,S,64]
  bf16*  Vt   = (bf16*)carve(nX * 2);             // [B,H,64,S]
  bf16*  At   = (bf16*)carve(nX * 2);             // [B*S, 1024]

  const int T = 256;
  cvt_f32_bf16<<<(int)((nX + T - 1) / T), T, 0, stream>>>(X,  Xb,  (int)nX);
  cvt_f32_bf16<<<(int)((nW + T - 1) / T), T, 0, stream>>>(wq, Wqb, (int)nW);
  cvt_f32_bf16<<<(int)((nW + T - 1) / T), T, 0, stream>>>(wk, Wkb, (int)nW);
  cvt_f32_bf16<<<(int)((nW + T - 1) / T), T, 0, stream>>>(wv, Wvb, (int)nW);
  cvt_f32_bf16<<<(int)((nW + T - 1) / T), T, 0, stream>>>(wo, Wob, (int)nW);

  const int gemmBlocks = (MTOT / 32) * (D_MODEL / 64) / 8;   // 512
  const int ropeN = MTOT * NUM_HEADS * (D_KH / 2);

  gemm_nt_bf16<<<gemmBlocks, 256, 0, stream>>>(Xb, Wqb, tmpF, MTOT, D_MODEL, D_MODEL);
  rope_repack<<<(ropeN + T - 1) / T, T, 0, stream>>>(tmpF, Qb);

  gemm_nt_bf16<<<gemmBlocks, 256, 0, stream>>>(Xb, Wkb, tmpF, MTOT, D_MODEL, D_MODEL);
  rope_repack<<<(ropeN + T - 1) / T, T, 0, stream>>>(tmpF, Kb);

  gemm_nt_bf16<<<gemmBlocks, 256, 0, stream>>>(Xb, Wvb, tmpF, MTOT, D_MODEL, D_MODEL);
  v_repack<<<(int)((nX + T - 1) / T), T, 0, stream>>>(tmpF, Vt);

  attn_flash<<<BATCH * NUM_HEADS * (SEQ / 32), 32, 0, stream>>>(Qb, Kb, Vt, At);

  gemm_nt_bf16<<<gemmBlocks, 256, 0, stream>>>(At, Wob, out, MTOT, D_MODEL, D_MODEL);
}
